// ctrnn_49787260895740
// MI455X (gfx1250) — matrix operand order repaired and checked by reading, not compiled, not run on hardware
//
#include <hip/hip_runtime.h>
#include <hip/hip_bf16.h>

#define DIM      2048
#define TIME     8192
#define DT       0.1f
#define NWG      16
#define BLOCK    256                 // 8 wave32 per workgroup
#define WAVES_PER_WG (BLOCK / 32)    // 8 -> 128 waves total = 128 output tiles
#define KCHUNKS  (DIM / 32)          // 64 K-chunks of 32

typedef __attribute__((ext_vector_type(16))) __bf16 v16bf;
typedef __attribute__((ext_vector_type(8)))  __bf16 v8bf;
typedef __attribute__((ext_vector_type(8)))  float  v8f;
typedef __attribute__((ext_vector_type(4)))  unsigned v4u;
typedef __attribute__((ext_vector_type(8)))  unsigned v8u;

union V16 { v16bf v; v8bf h[2]; };

// ---------------------------------------------------------------------------
// Prep: split W (f32) into bf16 hi/lo planes (row-major, so B fragments are
// contiguous 32B/lane loads), seed out[0] = initial, reset the grid barrier.
// ---------------------------------------------------------------------------
__global__ void ctrnn_prep(const float* __restrict__ W,
                           const float* __restrict__ initial,
                           float* __restrict__ out,
                           __bf16* __restrict__ whi,
                           __bf16* __restrict__ wlo,
                           int* __restrict__ bar) {
  const size_t i      = (size_t)blockIdx.x * blockDim.x + threadIdx.x;
  const size_t stride = (size_t)gridDim.x * blockDim.x;
  for (size_t idx = i; idx < (size_t)DIM * DIM; idx += stride) {
    float  w = W[idx];
    __bf16 h = (__bf16)w;
    whi[idx] = h;
    wlo[idx] = (__bf16)(w - (float)h);
  }
  if (i < DIM) out[i] = initial[i];
  if (i == 0) { bar[0] = 0; bar[1] = 0; }
}

// ---------------------------------------------------------------------------
// Persistent recurrent kernel.
// ---------------------------------------------------------------------------
__global__ void __launch_bounds__(BLOCK) ctrnn_main(
    const float*  __restrict__ inputs,   // [TIME][DIM]
    const float*  __restrict__ bias,     // [DIM]
    const float*  __restrict__ tau,      // [DIM]
    const __bf16* __restrict__ whi,      // [DIM][DIM]
    const __bf16* __restrict__ wlo,      // [DIM][DIM]
    float*        __restrict__ out,      // [TIME][DIM]
    int*          __restrict__ bar) {
  __shared__ float  sh_stage[DIM];       // TDM lands state t here (f32, 8KB)
  __shared__ __bf16 sh_hi[DIM];
  __shared__ __bf16 sh_lo[DIM];
  __shared__ __bf16 sh_zero[48];         // zero source for non-contributing A lanes

  const int tid  = threadIdx.x;
  const int lane = tid & 31;
  const int wave = tid >> 5;
  const int jt    = blockIdx.x * WAVES_PER_WG + wave;  // 0..127
  const int jbase = jt * 16;

  if (tid < 48) sh_zero[tid] = (__bf16)0.0f;

  // B fragment addressing: lane L -> W row (jbase + L%16), K half by L>=16.
  const int    nn     = lane & 15;
  const int    khalf  = lane >> 4;                      // 0 or 1 (x16 elements)
  const size_t rowoff = (size_t)(jbase + nn) * DIM;
  const v16bf* browhi = (const v16bf*)(whi + rowoff);
  const v16bf* browlo = (const v16bf*)(wlo + rowoff);

  // A fragment: row M=0 lives in lanes 0 and 16 only. Hoist the lane select
  // out of all loops: zero-lanes get a fixed sh_zero pointer and a 0 stride.
  const bool  a_lane  = (lane == 0) || (lane == 16);
  const int   akoff   = (lane == 16) ? 16 : 0;
  const char* ah_base = a_lane ? (const char*)&sh_hi[akoff] : (const char*)sh_zero;
  const char* al_base = a_lane ? (const char*)&sh_lo[akoff] : (const char*)sh_zero;
  const int   ainc    = a_lane ? 64 : 0;   // bytes per 32-element K chunk

  // Uniform pieces of the TDM descriptor (D#) for the 2048x1 f32 state row.
  const unsigned lds_off = (unsigned)(unsigned long long)(uintptr_t)&sh_stage[0];

  for (int t = 0; t < TIME - 1; ++t) {
    // --- Stage state t: TDM DMA global -> LDS, issued by wave 0 only. ------
    if (wave == 0) {
      unsigned long long ga =
          (unsigned long long)(uintptr_t)(out + (size_t)t * DIM);
      v4u g0;
      g0[0] = 1u;                                   // count=1 (valid D#)
      g0[1] = lds_off;                              // lds_addr
      g0[2] = (unsigned)(ga & 0xFFFFFFFFu);         // global_addr[31:0]
      g0[3] = (unsigned)((ga >> 32) & 0x01FFFFFFu)  // global_addr[56:32]
              | 0x80000000u;                        // type=2 ("image")
      v8u g1;
      g1[0] = 0x00020000u;   // data_size=2 (4B), no multicast/flags
      g1[1] = 0x08000000u;   // tensor_dim0 = 2048 (bits 63:48)
      g1[2] = 0x00010000u;   // tensor_dim1 = 1    (bits 95:80)
      g1[3] = 0x08000000u;   // tile_dim0   = 2048 (bits 127:112)
      g1[4] = 0x00000001u;   // tile_dim1   = 1    (bits 143:128)
      g1[5] = 2048u;         // tensor_dim0_stride[31:0]
      g1[6] = 0u;
      g1[7] = 0u;
      asm volatile("tensor_load_to_lds %0, %1" :: "s"(g0), "s"(g1) : "memory");
      __builtin_amdgcn_s_wait_tensorcnt(0);
    }
    __syncthreads();                     // sh_stage valid for whole block

    // Convert staged f32 state into split-bf16 planes (uniform trip count).
    for (int idx = tid; idx < DIM; idx += BLOCK) {
      float  v = sh_stage[idx];
      __bf16 h = (__bf16)v;
      sh_hi[idx] = h;
      sh_lo[idx] = (__bf16)(v - (float)h);
    }
    __syncthreads();

    // --- Matvec: D[0,n] = sum_k s[k] * W[jbase+n, k], split-bf16. ----------
    v8f c0 = {};
    v8f c1 = {};
    const char* ah = ah_base;
    const char* al = al_base;
    for (int kt = 0; kt < KCHUNKS; kt += 2) {
      const int kb0 = kt * 32;
      const int kb1 = kb0 + 32;

      V16 Ah0, Al0, Ah1, Al1;
      Ah0.h[0] = *(const v8bf*)(ah);          Ah0.h[1] = *(const v8bf*)(ah + 16);
      Al0.h[0] = *(const v8bf*)(al);          Al0.h[1] = *(const v8bf*)(al + 16);
      Ah1.h[0] = *(const v8bf*)(ah + ainc);   Ah1.h[1] = *(const v8bf*)(ah + ainc + 16);
      Al1.h[0] = *(const v8bf*)(al + ainc);   Al1.h[1] = *(const v8bf*)(al + ainc + 16);
      ah += 2 * ainc;
      al += 2 * ainc;

      v16bf Bh0 = browhi[(kb0 >> 4) + khalf];
      v16bf Bl0 = browlo[(kb0 >> 4) + khalf];
      v16bf Bh1 = browhi[(kb1 >> 4) + khalf];
      v16bf Bl1 = browlo[(kb1 >> 4) + khalf];

      c0 = __builtin_amdgcn_wmma_f32_16x16x32_bf16(false, Ah0.v, false, Bh0, (short)0, c0, false, false);
      c1 = __builtin_amdgcn_wmma_f32_16x16x32_bf16(false, Ah1.v, false, Bh1, (short)0, c1, false, false);
      c0 = __builtin_amdgcn_wmma_f32_16x16x32_bf16(false, Al0.v, false, Bh0, (short)0, c0, false, false);
      c1 = __builtin_amdgcn_wmma_f32_16x16x32_bf16(false, Al1.v, false, Bh1, (short)0, c1, false, false);
      c0 = __builtin_amdgcn_wmma_f32_16x16x32_bf16(false, Ah0.v, false, Bl0, (short)0, c0, false, false);
      c1 = __builtin_amdgcn_wmma_f32_16x16x32_bf16(false, Ah1.v, false, Bl1, (short)0, c1, false, false);
    }

    // Row M=0 of D sits in lanes 0..15, VGPR0: one output per lane.
    if (lane < 16) {
      const int j   = jbase + lane;
      float pre  = c0[0] + c1[0] + bias[j];
      float f    = 1.0f / (1.0f + __builtin_expf(-pre));
      float dtt  = DT / tau[j];
      float sj   = sh_stage[j];
      float snew = sj * (1.0f - dtt) + dtt * (f + inputs[(size_t)t * DIM + j]);
      out[(size_t)(t + 1) * DIM + j] = snew;
    }

    // --- Grid-wide sync: all writers of out[t+1] done before next step. ----
    __syncthreads();
    if (tid == 0) {
      __builtin_amdgcn_s_cluster_barrier();   // hardware NOP when not cluster-launched
      __threadfence();
      int g = __hip_atomic_load(&bar[1], __ATOMIC_RELAXED, __HIP_MEMORY_SCOPE_AGENT);
      int r = __hip_atomic_fetch_add(&bar[0], 1, __ATOMIC_ACQ_REL, __HIP_MEMORY_SCOPE_AGENT);
      if (r == NWG - 1) {
        __hip_atomic_store(&bar[0], 0, __ATOMIC_RELAXED, __HIP_MEMORY_SCOPE_AGENT);
        __hip_atomic_fetch_add(&bar[1], 1, __ATOMIC_RELEASE, __HIP_MEMORY_SCOPE_AGENT);
      } else {
        while (__hip_atomic_load(&bar[1], __ATOMIC_ACQUIRE, __HIP_MEMORY_SCOPE_AGENT) == g) {
          __builtin_amdgcn_s_sleep(2);
        }
      }
    }
    __syncthreads();
  }
}

// ---------------------------------------------------------------------------
extern "C" void kernel_launch(void* const* d_in, const int* in_sizes, int n_in,
                              void* d_out, int out_size, void* d_ws, size_t ws_size,
                              hipStream_t stream) {
  const float* inputs  = (const float*)d_in[0];  // [8192,1,2048]
  const float* W       = (const float*)d_in[1];  // [2048,2048]
  const float* bias    = (const float*)d_in[2];  // [2048]
  const float* tau     = (const float*)d_in[3];  // [1,2048]
  const float* initial = (const float*)d_in[4];  // [1,2048]
  float* out = (float*)d_out;                    // [8192,1,2048]

  __bf16* whi = (__bf16*)d_ws;
  __bf16* wlo = whi + (size_t)DIM * DIM;
  int*    bar = (int*)((char*)d_ws + 2 * (size_t)DIM * DIM * sizeof(__bf16));

  ctrnn_prep<<<2048, 256, 0, stream>>>(W, initial, out, whi, wlo, bar);
  ctrnn_main<<<NWG, BLOCK, 0, stream>>>(inputs, bias, tau, whi, wlo, out, bar);
}
